// MultiHeadAttention_6184752906430
// MI455X (gfx1250) — matrix operand order repaired and checked by reading, not compiled, not run on hardware
//
#include <hip/hip_runtime.h>
#include <hip/hip_bf16.h>

// ---------------------------------------------------------------------------
// GQA attention block for MI455X (gfx1250, wave32, WMMA + TDM).
// Compute-bound (~56 GFLOP vs ~40MB traffic) -> all matrix math through
// V_WMMA_F32_16X16X32_BF16; tile staging through the Tensor Data Mover
// (TENSOR_LOAD_TO_LDS, TENSORcnt). Weights/V pre-transposed once so every
// WMMA fragment load is a contiguous (b128-fusable) LDS read.
// ---------------------------------------------------------------------------

#define BB 2
#define TT 2048
#define CC 1024
#define NQH 16
#define GROUP 4
#define HKV 4
#define HS 64
#define OUT_DIM (CC + 2 * (CC / GROUP))   // 1536
#define MROWS (BB * TT)                    // 4096
#define SCALE 0.125f                       // 1/sqrt(64)
#define NEG_INF (-3.0e38f)

typedef __attribute__((ext_vector_type(16))) __bf16 v16bf;
typedef __attribute__((ext_vector_type(8)))  float  v8f;

union FragB16 { v16bf v; unsigned u[8]; };

#ifndef __has_builtin
#define __has_builtin(x) 0
#endif
#if __has_builtin(__builtin_amdgcn_tensor_load_to_lds)
#define HAVE_TDM 1
#else
#define HAVE_TDM 0
#endif

#if HAVE_TDM
typedef unsigned u32x4 __attribute__((ext_vector_type(4)));
typedef int      i32x4 __attribute__((ext_vector_type(4)));
typedef int      i32x8 __attribute__((ext_vector_type(8)));

// One 2D TDM load: tile [tile_d1 rows x tile_d0 elems] of bf16 from a
// row-major tensor (row stride stride0 elems) into LDS at lds_byte_addr,
// with LDS row padding (pad_interval_code, pad_amount_code).
// D# fields per CDNA5 ISA.
__device__ __forceinline__ void tdm_load_2d(
    unsigned lds_byte_addr, const __bf16* gsrc,
    unsigned tensor_d0, unsigned tensor_d1, unsigned stride0,
    unsigned tile_d0, unsigned tile_d1,
    unsigned pad_interval_code, unsigned pad_amount_code) {
  unsigned long long ga = (unsigned long long)(size_t)gsrc;
  u32x4 g0;
  g0[0] = 1u;                                   // count=1 (valid), user mode
  g0[1] = lds_byte_addr;                        // lds_addr
  g0[2] = (unsigned)(ga & 0xFFFFFFFFu);         // global_addr[31:0]
  g0[3] = (unsigned)((ga >> 32) & 0x01FFFFFFu)  // global_addr[56:32]
          | (2u << 30);                         // type = 2 ("image")
  i32x8 g1;
  g1[0] = (int)((1u << 16)                      // data_size = 1 -> 2 bytes
                | (1u << 20)                    // pad_enable
                | (pad_interval_code << 22)
                | (pad_amount_code << 25));
  g1[1] = (int)((tensor_d0 & 0xFFFFu) << 16);
  g1[2] = (int)((tensor_d0 >> 16) | ((tensor_d1 & 0xFFFFu) << 16));
  g1[3] = (int)((tensor_d1 >> 16) | (tile_d0 << 16));
  g1[4] = (int)(tile_d1 & 0xFFFFu);
  g1[5] = (int)stride0;
  g1[6] = 0;
  g1[7] = 0;
  i32x4 g2 = {0, 0, 0, 0};
  i32x4 g3 = {0, 0, 0, 0};
#if __clang_major__ >= 23
  i32x8 g4 = {0, 0, 0, 0, 0, 0, 0, 0};
  __builtin_amdgcn_tensor_load_to_lds(g0, g1, g2, g3, g4, 0);
#else
  __builtin_amdgcn_tensor_load_to_lds(g0, g1, g2, g3, 0);
#endif
}
#endif  // HAVE_TDM

// ---------------------------------------------------------------------------
// f32 -> bf16 elementwise convert
// ---------------------------------------------------------------------------
__global__ void f32_to_bf16_kernel(const float* __restrict__ src,
                                   __bf16* __restrict__ dst, int n) {
  int i = blockIdx.x * blockDim.x + threadIdx.x;
  if (i < n) dst[i] = (__bf16)src[i];
}

// f32 src[K][N] -> bf16 dst[N][K] (one-time weight transpose)
__global__ void f32_to_bf16_transpose_kernel(const float* __restrict__ src,
                                             __bf16* __restrict__ dst,
                                             int K, int N) {
  int i = blockIdx.x * blockDim.x + threadIdx.x;
  if (i < N * K) {
    int k = i % K;
    int n = i / K;
    dst[i] = (__bf16)src[(size_t)k * N + n];
  }
}

// ---------------------------------------------------------------------------
// Tiled WMMA GEMM: D[M,N] = A[M,K](bf16) * Bt^T + bias[N], Bt is [N][K] bf16.
// Block tile 128x64, 256 threads = 8 waves, each wave a 32x32 region
// (2x2 WMMA 16x16 tiles). K-step = 32. Tiles staged by TDM when available.
// Both A and Bt tiles are k-contiguous -> all fragment loads contiguous.
// ---------------------------------------------------------------------------
__global__ __launch_bounds__(256) void wmma_gemm_bias(
    const __bf16* __restrict__ A, const __bf16* __restrict__ Bt,
    const float* __restrict__ bias, float* __restrict__ D,
    int M, int N, int K) {
  __shared__ __bf16 As[128][48];  // [m][k]: 32 elems + 16 pad (16 DW + 8 DW)
  __shared__ __bf16 Bs[64][48];   // [n][k]: 32 elems + 16 pad

  const int tid = threadIdx.x;
  const int lane = tid & 31;
  const int wid = tid >> 5;
  const int ln = lane & 15;
  const int hi = lane >> 4;
  const int wm = wid >> 1;   // 0..3 : 32-row strip
  const int wn = wid & 1;    // 0..1 : 32-col strip
  const int bm = blockIdx.x * 128;
  const int bn = blockIdx.y * 64;

  v8f acc[2][2] = {};

  for (int k0 = 0; k0 < K; k0 += 32) {
    __syncthreads();
#if HAVE_TDM
    if (wid == 0) {
      tdm_load_2d((unsigned)(size_t)&As[0][0], A + (size_t)bm * K + k0,
                  (unsigned)K, (unsigned)M, (unsigned)K, 32u, 128u, 3u, 7u);
      tdm_load_2d((unsigned)(size_t)&Bs[0][0], Bt + (size_t)bn * K + k0,
                  (unsigned)K, (unsigned)N, (unsigned)K, 32u, 64u, 3u, 7u);
      __builtin_amdgcn_s_wait_tensorcnt(0);
    }
#else
    for (int it = 0; it < 2; ++it) {  // A tile: 128x32
      int idx = (tid + it * 256) * 8;
      int r = idx >> 5, c = idx & 31;
      *(uint4*)&As[r][c] = *(const uint4*)(A + (size_t)(bm + r) * K + k0 + c);
    }
    {  // Bt tile: 64x32
      int idx = tid * 8;
      int r = idx >> 5, c = idx & 31;
      *(uint4*)&Bs[r][c] = *(const uint4*)(Bt + (size_t)(bn + r) * K + k0 + c);
    }
#endif
    __syncthreads();

    // A fragments: ISA 16-bit A 16x32 layout (kk = (v&4)<<2 | hi*8 | (v&3)*2)
    FragB16 af[2], bfr[2];
    for (int i = 0; i < 2; ++i) {
      int m = wm * 32 + i * 16 + ln;
      for (int v = 0; v < 8; ++v) {
        int kk = ((v & 4) << 2) + hi * 8 + ((v & 3) << 1);
        af[i].u[v] = *(const unsigned*)&As[m][kk];
      }
    }
    // B fragments: same k order as A (col = ln, kk = (v&4)<<2 | hi*8 | (v&3)*2);
    // Bs is [n][k] so pairs are contiguous.
    for (int j = 0; j < 2; ++j) {
      int n = wn * 32 + j * 16 + ln;
      for (int v = 0; v < 8; ++v) {
        int kk = ((v & 4) << 2) + hi * 8 + ((v & 3) << 1);
        bfr[j].u[v] = *(const unsigned*)&Bs[n][kk];
      }
    }
    for (int i = 0; i < 2; ++i)
      for (int j = 0; j < 2; ++j)
        acc[i][j] = __builtin_amdgcn_wmma_f32_16x16x32_bf16(
            false, af[i].v, false, bfr[j].v, (short)0, acc[i][j], false, false);
  }

  // C/D layout: row = 8*hi + r, col = ln
  for (int i = 0; i < 2; ++i)
    for (int j = 0; j < 2; ++j)
      for (int r = 0; r < 8; ++r) {
        int row = bm + wm * 32 + i * 16 + 8 * hi + r;
        int col = bn + wn * 32 + j * 16 + ln;
        D[(size_t)row * N + col] = acc[i][j][r] + bias[col];
      }
}

// ---------------------------------------------------------------------------
// RoPE + scatter: qkv f32 -> Qb [b,h,g,t,hs] bf16 (roped),
//                 Kb [b,h,t,hs] bf16 (roped), Vt [b,h,hs,t] bf16 (transposed)
// ---------------------------------------------------------------------------
__global__ void rope_scatter_kernel(const float* __restrict__ qkv,
                                    __bf16* __restrict__ Qb,
                                    __bf16* __restrict__ Kb,
                                    __bf16* __restrict__ Vt) {
  const int NQ = BB * TT * CC / 2;
  const int NKV = BB * TT * HKV * HS / 2;
  size_t i = (size_t)blockIdx.x * blockDim.x + threadIdx.x;
  if (i < (size_t)NQ) {
    int j = (int)(i & 31);
    size_t r = i >> 5;
    int h = (int)(r % HKV); r /= HKV;
    int g = (int)(r % GROUP); r /= GROUP;
    int t = (int)(r % TT);
    int b = (int)(r / TT);
    int c0 = ((g * HKV) + h) * HS + 2 * j;
    const float* src = qkv + (size_t)(b * TT + t) * OUT_DIM + c0;
    float x1 = src[0], x2 = src[1];
    float inv = __expf(-((float)j / 32.0f) * 9.21034037f);  // 10000^(-j/32)
    float s, c;
    __sincosf((float)t * inv, &s, &c);
    __bf16* dst = Qb + ((((size_t)b * HKV + h) * GROUP + g) * TT + t) * HS + 2 * j;
    dst[0] = (__bf16)(x1 * c - x2 * s);
    dst[1] = (__bf16)(x1 * s + x2 * c);
  } else if (i < (size_t)(NQ + NKV)) {
    size_t q = i - NQ;
    int j = (int)(q & 31);
    size_t r = q >> 5;
    int h = (int)(r % HKV); r /= HKV;
    int t = (int)(r % TT);
    int b = (int)(r / TT);
    const float* base = qkv + (size_t)(b * TT + t) * OUT_DIM;
    int ck = CC + h * HS + 2 * j;
    int cv = CC + (HKV + h) * HS + 2 * j;
    float x1 = base[ck], x2 = base[ck + 1];
    float inv = __expf(-((float)j / 32.0f) * 9.21034037f);
    float s, c;
    __sincosf((float)t * inv, &s, &c);
    size_t ok = (((size_t)b * HKV + h) * TT + t) * HS + 2 * j;
    Kb[ok] = (__bf16)(x1 * c - x2 * s);
    Kb[ok + 1] = (__bf16)(x1 * s + x2 * c);
    // V transposed: [b,h,hs,t]
    size_t ov = (((size_t)b * HKV + h) * HS + 2 * j) * TT + t;
    Vt[ov] = (__bf16)base[cv];
    Vt[ov + TT] = (__bf16)base[cv + 1];
  }
}

// ---------------------------------------------------------------------------
// Flash-style causal attention.
// Block: 256 threads = 8 waves, 128 q-rows of one (b, h, g). Each wave owns a
// 16-row strip. Streams 64-key blocks (TDM-staged) with online softmax; WMMA
// for QK^T and PV. V is pre-transposed so PV B-fragments are contiguous.
// ---------------------------------------------------------------------------
__global__ __launch_bounds__(256) void attn_kernel(
    const __bf16* __restrict__ Qb, const __bf16* __restrict__ Kb,
    const __bf16* __restrict__ Vt, __bf16* __restrict__ Ob) {
  __shared__ __bf16 Ks[64][72];        // [key][hs]
  __shared__ __bf16 Vs[64][72];        // [hs][key]
  __shared__ __bf16 Ps[8][16][72];     // per-wave P staging (C->A layout)

  const int tid = threadIdx.x;
  const int lane = tid & 31;
  const int wid = tid >> 5;
  const int ln = lane & 15;
  const int hi = lane >> 4;

  const int qt = blockIdx.x;
  const int g = blockIdx.y & (GROUP - 1);
  const int h = blockIdx.y >> 2;
  const int b = blockIdx.z;

  const int qbase = qt * 128 + wid * 16;

  const __bf16* qptr = Qb + ((((size_t)b * HKV + h) * GROUP + g) * TT) * HS;
  FragB16 qa[2];
  for (int f = 0; f < 2; ++f)
    for (int v = 0; v < 8; ++v) {
      int kk = f * 32 + ((v & 4) << 2) + hi * 8 + ((v & 3) << 1);
      qa[f].u[v] = *(const unsigned*)(qptr + (size_t)(qbase + ln) * HS + kk);
    }

  const __bf16* kptr = Kb + (((size_t)b * HKV + h) * TT) * HS;
  const __bf16* vptr = Vt + (((size_t)b * HKV + h) * HS) * TT;

  float mrow[8], lrow[8];
  for (int r = 0; r < 8; ++r) { mrow[r] = NEG_INF; lrow[r] = 0.0f; }
  v8f oacc[4] = {};

  const int nkb = 2 * qt + 2;
  for (int kb = 0; kb < nkb; ++kb) {
    __syncthreads();
#if HAVE_TDM
    if (wid == 0) {
      // K tile: 64 keys x 64 hs, row stride HS; LDS row = 32 DW + 4 DW pad
      tdm_load_2d((unsigned)(size_t)&Ks[0][0], kptr + (size_t)kb * 64 * HS,
                  (unsigned)HS, (unsigned)TT, (unsigned)HS, 64u, 64u, 4u, 3u);
      // V tile (transposed layout): 64 hs-rows x 64 keys, row stride TT
      tdm_load_2d((unsigned)(size_t)&Vs[0][0], vptr + (size_t)kb * 64,
                  (unsigned)TT, (unsigned)HS, (unsigned)TT, 64u, 64u, 4u, 3u);
      __builtin_amdgcn_s_wait_tensorcnt(0);
    }
#else
    for (int it = 0; it < 2; ++it) {
      int idx = (tid + it * 256) * 8;
      int r = idx >> 6, c = idx & 63;
      *(uint4*)&Ks[r][c] = *(const uint4*)(kptr + (size_t)(kb * 64 + r) * HS + c);
      *(uint4*)&Vs[r][c] = *(const uint4*)(vptr + (size_t)r * TT + kb * 64 + c);
    }
#endif
    __syncthreads();

    if (kb * 64 <= qbase + 15) {  // strip not fully masked
      // ---- S = Q * K^T ---- (B frag: col = key, k = hs in A's k order; pairs contiguous)
      v8f sacc[4] = {};
      for (int f = 0; f < 2; ++f)
        for (int t = 0; t < 4; ++t) {
          FragB16 bf_;
          for (int v = 0; v < 8; ++v) {
            int kk = f * 32 + ((v & 4) << 2) + hi * 8 + ((v & 3) << 1);
            bf_.u[v] = *(const unsigned*)&Ks[t * 16 + ln][kk];
          }
          sacc[t] = __builtin_amdgcn_wmma_f32_16x16x32_bf16(
              false, qa[f].v, false, bf_.v, (short)0, sacc[t], false, false);
        }

      // ---- scale, causal mask, online softmax ----
      float p[4][8], alpha[8];
      for (int r = 0; r < 8; ++r) {
        int row = qbase + 8 * hi + r;
        float mx = NEG_INF;
        for (int t = 0; t < 4; ++t) {
          int col = kb * 64 + t * 16 + ln;
          float s = sacc[t][r] * SCALE;
          if (col > row) s = NEG_INF;
          p[t][r] = s;
          mx = fmaxf(mx, s);
        }
        mx = fmaxf(mx, __shfl_xor(mx, 1, 32));
        mx = fmaxf(mx, __shfl_xor(mx, 2, 32));
        mx = fmaxf(mx, __shfl_xor(mx, 4, 32));
        mx = fmaxf(mx, __shfl_xor(mx, 8, 32));
        float mnew = fmaxf(mrow[r], mx);
        alpha[r] = __expf(mrow[r] - mnew);
        float ls = 0.0f;
        for (int t = 0; t < 4; ++t) {
          float e = __expf(p[t][r] - mnew);
          p[t][r] = e;
          ls += e;
        }
        ls += __shfl_xor(ls, 1, 32);
        ls += __shfl_xor(ls, 2, 32);
        ls += __shfl_xor(ls, 4, 32);
        ls += __shfl_xor(ls, 8, 32);
        lrow[r] = lrow[r] * alpha[r] + ls;
        mrow[r] = mnew;
      }

      // ---- rescale O, stage P ----
      for (int t = 0; t < 4; ++t)
        for (int r = 0; r < 8; ++r) {
          oacc[t][r] *= alpha[r];
          Ps[wid][8 * hi + r][t * 16 + ln] = (__bf16)p[t][r];
        }

      // ---- O += P * V ---- (Vs[hs][key]: pairs along key contiguous)
      for (int f = 0; f < 2; ++f) {
        FragB16 pa;
        for (int v = 0; v < 8; ++v) {
          int kk = f * 32 + ((v & 4) << 2) + hi * 8 + ((v & 3) << 1);
          pa.u[v] = *(const unsigned*)&Ps[wid][ln][kk];
        }
        for (int t = 0; t < 4; ++t) {
          FragB16 vb_;
          for (int v = 0; v < 8; ++v) {
            int kk = f * 32 + ((v & 4) << 2) + hi * 8 + ((v & 3) << 1);
            vb_.u[v] = *(const unsigned*)&Vs[t * 16 + ln][kk];
          }
          oacc[t] = __builtin_amdgcn_wmma_f32_16x16x32_bf16(
              false, pa.v, false, vb_.v, (short)0, oacc[t], false, false);
        }
      }
    }
  }

  // ---- normalize and write to [b, t, (g*HKV+h)*HS + hs] as bf16 ----
  const int cbase = (g * HKV + h) * HS;
  for (int r = 0; r < 8; ++r) {
    int row = qbase + 8 * hi + r;
    float inv = 1.0f / lrow[r];
    for (int t = 0; t < 4; ++t)
      Ob[((size_t)b * TT + row) * CC + cbase + t * 16 + ln] =
          (__bf16)(oacc[t][r] * inv);
  }
}

// ---------------------------------------------------------------------------
// Host launcher
// ---------------------------------------------------------------------------
extern "C" void kernel_launch(void* const* d_in, const int* in_sizes, int n_in,
                              void* d_out, int out_size, void* d_ws, size_t ws_size,
                              hipStream_t stream) {
  (void)in_sizes; (void)n_in; (void)out_size; (void)ws_size;
  const float* input = (const float*)d_in[0];
  const float* W_attn = (const float*)d_in[1];
  const float* b_attn = (const float*)d_in[2];
  const float* W_final = (const float*)d_in[3];
  const float* b_final = (const float*)d_in[4];
  float* out = (float*)d_out;
  char* ws = (char*)d_ws;

  size_t off = 0;
  __bf16* inBf = (__bf16*)(ws + off);     off += (size_t)MROWS * CC * 2;
  __bf16* wAttnT = (__bf16*)(ws + off);   off += (size_t)CC * OUT_DIM * 2;  // [N][K]
  __bf16* wFinT = (__bf16*)(ws + off);    off += (size_t)CC * CC * 2;       // [N][K]
  float*  qkv = (float*)(ws + off);       off += (size_t)MROWS * OUT_DIM * 4;
  __bf16* Qb = (__bf16*)(ws + off);       off += (size_t)MROWS * CC * 2;
  __bf16* Kb = (__bf16*)(ws + off);       off += (size_t)BB * HKV * TT * HS * 2;
  __bf16* Vt = (__bf16*)(ws + off);       off += (size_t)BB * HKV * TT * HS * 2;
  __bf16* Ob = (__bf16*)(ws + off);       off += (size_t)MROWS * CC * 2;

  {
    int n1 = MROWS * CC;
    f32_to_bf16_kernel<<<(n1 + 255) / 256, 256, 0, stream>>>(input, inBf, n1);
    int n2 = CC * OUT_DIM;
    f32_to_bf16_transpose_kernel<<<(n2 + 255) / 256, 256, 0, stream>>>(
        W_attn, wAttnT, CC, OUT_DIM);
    int n3 = CC * CC;
    f32_to_bf16_transpose_kernel<<<(n3 + 255) / 256, 256, 0, stream>>>(
        W_final, wFinT, CC, CC);
  }
  {
    dim3 grid(MROWS / 128, OUT_DIM / 64);
    wmma_gemm_bias<<<grid, 256, 0, stream>>>(inBf, wAttnT, b_attn, qkv,
                                             MROWS, OUT_DIM, CC);
  }
  {
    size_t ntot = (size_t)(BB * TT * CC / 2) + (size_t)(BB * TT * HKV * HS / 2);
    int blocks = (int)((ntot + 255) / 256);
    rope_scatter_kernel<<<blocks, 256, 0, stream>>>(qkv, Qb, Kb, Vt);
  }
  {
    dim3 grid(TT / 128, HKV * GROUP, BB);
    attn_kernel<<<grid, 256, 0, stream>>>(Qb, Kb, Vt, Ob);
  }
  {
    dim3 grid(MROWS / 128, CC / 64);
    wmma_gemm_bias<<<grid, 256, 0, stream>>>(Ob, wFinT, b_final, out,
                                             MROWS, CC, CC);
  }
}
